// TaskEmbedding_17076789968986
// MI455X (gfx1250) — matrix operand order repaired and checked by reading, not compiled, not run on hardware
//
#include <hip/hip_runtime.h>
#include <hip/hip_bf16.h>
#include <math.h>

// ============================================================================
// TaskEmbedding for MI455X (gfx1250, wave32, WMMA).
//
// fp32 GEMMs emulated on v_wmma_f32_16x16x32_f16 via f16 hi/lo split
// (3 WMMAs per tile-K step, ~22 mantissa bits). fp32->f16(hi,lo) conversion
// happens ONCE per block per K-step while staging to LDS; LDS holds packed
// half2 words so fragments are plain LDS dword loads (b128 for A / NT-B) in
// contiguous k = 16*hi + s order. Cosine scaling fused into NT epilogue; softmax
// in-place in the G output regions; NN GEMM writes straight into the
// concatenated outputs.
// ============================================================================

typedef __attribute__((ext_vector_type(16))) _Float16 v16h;
typedef __attribute__((ext_vector_type(8)))  float    v8f;

#define COS_EPS 1e-8f

// ---------------- tiling ----------------
#define BM   128           // block tile M
#define BN   128           // block tile N
#define BK   32            // K step == WMMA K
#define KH   16            // BK/2 packed half2 words per row
#define LDT  20            // padded dword row stride (16 used): 16B aligned,
                           // 20*lm mod 64 distinct -> bank-conflict-free b128
#define LDB2 132           // NN B tile: [kpair][n] dword stride (128 + 4 pad)

// ---------------------------------------------------------------------------
// helpers
// ---------------------------------------------------------------------------
__device__ __forceinline__ void cvt_pair(float x, float y,
                                         unsigned int& whi, unsigned int& wlo) {
  union { _Float16 h[2]; unsigned int u; } H, L;
  const _Float16 hx = (_Float16)x, hy = (_Float16)y;
  H.h[0] = hx;
  H.h[1] = hy;
  L.h[0] = (_Float16)(x - (float)hx);
  L.h[1] = (_Float16)(y - (float)hy);
  whi = H.u;
  wlo = L.u;
}

// A fragment: 16x32 f16, lane(0-15)->M, VGPR v K-pair hi*16+2v
// => dword index hi*8 + v: two contiguous uint4 loads.
__device__ __forceinline__ v16h frag_A(const unsigned int* __restrict__ T,
                                       int row, int hi) {
  union { v16h h; unsigned int u[8]; } F;
  const unsigned int* p = T + row * LDT + hi * 8;
  const uint4 q0 = *(const uint4*)(p);
  const uint4 q1 = *(const uint4*)(p + 4);
  F.u[0] = q0.x; F.u[1] = q0.y; F.u[2] = q0.z; F.u[3] = q0.w;
  F.u[4] = q1.x; F.u[5] = q1.y; F.u[6] = q1.z; F.u[7] = q1.w;
  return F.h;
}

// B fragment (NT tile stored [n][kpair]): VGPR v K-pair hi*16+2v
// => dword index hi*8 + v: two contiguous uint4 loads.
__device__ __forceinline__ v16h frag_Bnt(const unsigned int* __restrict__ T,
                                         int row, int hi) {
  union { v16h h; unsigned int u[8]; } F;
  const unsigned int* p = T + row * LDT + hi * 8;
  const uint4 q0 = *(const uint4*)(p);
  const uint4 q1 = *(const uint4*)(p + 4);
  F.u[0] = q0.x; F.u[1] = q0.y; F.u[2] = q0.z; F.u[3] = q0.w;
  F.u[4] = q1.x; F.u[5] = q1.y; F.u[6] = q1.z; F.u[7] = q1.w;
  return F.h;
}

// B fragment (NN tile stored [kpair][n]): word kp = hi*8+v at column n.
__device__ __forceinline__ v16h frag_Bnn(const unsigned int* __restrict__ T,
                                         int n, int hi) {
  union { v16h h; unsigned int u[8]; } F;
#pragma unroll
  for (int v = 0; v < 8; ++v) F.u[v] = T[(hi * 8 + v) * LDB2 + n];
  return F.h;
}

// fp32-emulated MAC: c += (ah+al)*(bh+bl), dropping lo*lo
__device__ __forceinline__ v8f wmma3(v16h ah, v16h al, v16h bh, v16h bl, v8f c) {
  c = __builtin_amdgcn_wmma_f32_16x16x32_f16(false, ah, false, bh, (short)0, c, false, false);
  c = __builtin_amdgcn_wmma_f32_16x16x32_f16(false, al, false, bh, (short)0, c, false, false);
  c = __builtin_amdgcn_wmma_f32_16x16x32_f16(false, ah, false, bl, (short)0, c, false, false);
  return c;
}

// ---------------------------------------------------------------------------
// row inverse norms: inv[row] = 1 / max(||x_row||, eps)
// ---------------------------------------------------------------------------
__global__ __launch_bounds__(256) void rownorm_kernel(
    const float* __restrict__ x, float* __restrict__ inv, int d) {
  const int row = blockIdx.x;
  const float* p = x + (size_t)row * d;
  float s = 0.f;
  for (int i = threadIdx.x; i < d; i += 256) { float v = p[i]; s += v * v; }
  __shared__ float red[256];
  red[threadIdx.x] = s;
  __syncthreads();
  for (int off = 128; off > 0; off >>= 1) {
    if (threadIdx.x < off) red[threadIdx.x] += red[threadIdx.x + off];
    __syncthreads();
  }
  if (threadIdx.x == 0) inv[row] = 1.f / fmaxf(sqrtf(red[0]), COS_EPS);
}

// ---------------------------------------------------------------------------
// in-place row softmax, n == 1024, one 256-thread block per row
// ---------------------------------------------------------------------------
__global__ __launch_bounds__(256) void softmax_rows(float* __restrict__ x) {
  float* p = x + (size_t)blockIdx.x * 1024;
  __shared__ float red[256];
  float vals[4];
  float lmax = -3.4e38f;
#pragma unroll
  for (int i = 0; i < 4; ++i) {
    vals[i] = p[threadIdx.x + i * 256];
    lmax = fmaxf(lmax, vals[i]);
  }
  red[threadIdx.x] = lmax;
  __syncthreads();
  for (int off = 128; off > 0; off >>= 1) {
    if (threadIdx.x < off) red[threadIdx.x] = fmaxf(red[threadIdx.x], red[threadIdx.x + off]);
    __syncthreads();
  }
  const float rmax = red[0];
  __syncthreads();
  float ls = 0.f;
#pragma unroll
  for (int i = 0; i < 4; ++i) { vals[i] = __expf(vals[i] - rmax); ls += vals[i]; }
  red[threadIdx.x] = ls;
  __syncthreads();
  for (int off = 128; off > 0; off >>= 1) {
    if (threadIdx.x < off) red[threadIdx.x] += red[threadIdx.x + off];
    __syncthreads();
  }
  const float rinv = 1.f / red[0];
#pragma unroll
  for (int i = 0; i < 4; ++i) p[threadIdx.x + i * 256] = vals[i] * rinv;
}

// ---------------------------------------------------------------------------
// NT GEMM + cosine scaling: C[b,m,n] = (A[b,m,:].Bm[b,n,:]) * sA[b,m]*sB[b,n]
// grid (N/BN, M/BM, B), 256 threads = 8 waves (4 along M x 2 along N),
// wave tile 32x64 = 2x4 WMMA accumulators.
// ---------------------------------------------------------------------------
__global__ __launch_bounds__(256) void gemm_nt_cos(
    const float* __restrict__ A, const float* __restrict__ Bm,
    float* __restrict__ C,
    const float* __restrict__ sA, const float* __restrict__ sB,
    int M, int N, int Kd) {
  __shared__ unsigned int Ah[BM * LDT], Al[BM * LDT];
  __shared__ unsigned int Bh[BN * LDT], Bl[BN * LDT];

  const int b  = blockIdx.z;
  const int m0 = blockIdx.y * BM;
  const int n0 = blockIdx.x * BN;
  const float* Ab = A  + (size_t)b * M * Kd;
  const float* Bb = Bm + (size_t)b * N * Kd;

  const int tid   = threadIdx.x;
  const int wave  = tid >> 5;
  const int lane  = tid & 31;
  const int lm    = lane & 15;
  const int hi    = lane >> 4;
  const int waveM = (wave >> 1) * 32;  // 4 waves along M
  const int waveN = (wave & 1) * 64;   // 2 waves along N

  v8f acc[2][4];
#pragma unroll
  for (int i = 0; i < 2; ++i)
#pragma unroll
    for (int j = 0; j < 4; ++j) acc[i][j] = (v8f){0, 0, 0, 0, 0, 0, 0, 0};

  for (int k0 = 0; k0 < Kd; k0 += BK) {
    // stage A tile: BM x BK f32 -> packed half2 hi/lo (4 float4 per thread)
#pragma unroll
    for (int i = tid; i < BM * 8; i += 256) {
      const int r = i >> 3, c4 = i & 7;
      const float* src = Ab + (size_t)(m0 + r) * Kd + k0 + c4 * 4;
      if (k0 + BK < Kd) __builtin_prefetch(src + BK, 0, 1);
      const float4 v = *(const float4*)src;
      unsigned int h0, l0, h1, l1;
      cvt_pair(v.x, v.y, h0, l0);
      cvt_pair(v.z, v.w, h1, l1);
      *(uint2*)&Ah[r * LDT + c4 * 2] = make_uint2(h0, h1);
      *(uint2*)&Al[r * LDT + c4 * 2] = make_uint2(l0, l1);
    }
    // stage B rows (same packing, tile stored [n][kpair])
#pragma unroll
    for (int i = tid; i < BN * 8; i += 256) {
      const int r = i >> 3, c4 = i & 7;
      const float* src = Bb + (size_t)(n0 + r) * Kd + k0 + c4 * 4;
      if (k0 + BK < Kd) __builtin_prefetch(src + BK, 0, 1);
      const float4 v = *(const float4*)src;
      unsigned int h0, l0, h1, l1;
      cvt_pair(v.x, v.y, h0, l0);
      cvt_pair(v.z, v.w, h1, l1);
      *(uint2*)&Bh[r * LDT + c4 * 2] = make_uint2(h0, h1);
      *(uint2*)&Bl[r * LDT + c4 * 2] = make_uint2(l0, l1);
    }
    __syncthreads();

    v16h ah[2], al[2];
#pragma unroll
    for (int t = 0; t < 2; ++t) {
      ah[t] = frag_A(Ah, waveM + t * 16 + lm, hi);
      al[t] = frag_A(Al, waveM + t * 16 + lm, hi);
    }
#pragma unroll
    for (int tj = 0; tj < 4; ++tj) {
      const v16h bh = frag_Bnt(Bh, waveN + tj * 16 + lm, hi);
      const v16h bl = frag_Bnt(Bl, waveN + tj * 16 + lm, hi);
#pragma unroll
      for (int ti = 0; ti < 2; ++ti)
        acc[ti][tj] = wmma3(ah[ti], al[ti], bh, bl, acc[ti][tj]);
    }
    __syncthreads();
  }

  // epilogue with cosine scaling (D VGPR i -> row i + hi*8; col = lane%16)
  float* Cb = C + (size_t)b * M * N;
  const float* sAb = sA + (size_t)b * M;
  const float* sBb = sB + (size_t)b * N;
#pragma unroll
  for (int ti = 0; ti < 2; ++ti)
#pragma unroll
    for (int tj = 0; tj < 4; ++tj) {
      const int col = n0 + waveN + tj * 16 + lm;
      const float scol = sBb[col];
#pragma unroll
      for (int i = 0; i < 8; ++i) {
        const int row = m0 + waveM + ti * 16 + hi * 8 + i;
        Cb[(size_t)row * N + col] = acc[ti][tj][i] * sAb[row] * scol;
      }
    }
}

// ---------------------------------------------------------------------------
// NN GEMM into concatenated output right half:
//   Out[b, m, N + n] = sum_k G[b,m,k] * E[b,k,n]
// grid (N/BN, M/BM, B); N = 1024 (emb dim), K = 1024 (support count)
// ---------------------------------------------------------------------------
__global__ __launch_bounds__(256) void gemm_nn_task(
    const float* __restrict__ G, const float* __restrict__ E,
    float* __restrict__ Out, int M, int N, int Kd) {
  __shared__ unsigned int Ah[BM * LDT], Al[BM * LDT];
  __shared__ unsigned int Bh[KH * LDB2], Bl[KH * LDB2];

  const int b  = blockIdx.z;
  const int m0 = blockIdx.y * BM;
  const int n0 = blockIdx.x * BN;
  const float* Gb = G + (size_t)b * M * Kd;
  const float* Eb = E + (size_t)b * Kd * N;

  const int tid   = threadIdx.x;
  const int wave  = tid >> 5;
  const int lane  = tid & 31;
  const int lm    = lane & 15;
  const int hi    = lane >> 4;
  const int waveM = (wave >> 1) * 32;
  const int waveN = (wave & 1) * 64;

  v8f acc[2][4];
#pragma unroll
  for (int i = 0; i < 2; ++i)
#pragma unroll
    for (int j = 0; j < 4; ++j) acc[i][j] = (v8f){0, 0, 0, 0, 0, 0, 0, 0};

  for (int k0 = 0; k0 < Kd; k0 += BK) {
    // stage A tile from G (row-major along K)
#pragma unroll
    for (int i = tid; i < BM * 8; i += 256) {
      const int r = i >> 3, c4 = i & 7;
      const float* src = Gb + (size_t)(m0 + r) * Kd + k0 + c4 * 4;
      if (k0 + BK < Kd) __builtin_prefetch(src + BK, 0, 1);
      const float4 v = *(const float4*)src;
      unsigned int h0, l0, h1, l1;
      cvt_pair(v.x, v.y, h0, l0);
      cvt_pair(v.z, v.w, h1, l1);
      *(uint2*)&Ah[r * LDT + c4 * 2] = make_uint2(h0, h1);
      *(uint2*)&Al[r * LDT + c4 * 2] = make_uint2(l0, l1);
    }
    // stage B tile from E rows k0..k0+31: pack K-pairs across two rows,
    // store [kpair][n] (coalesced b128 stores; 2 items/thread)
#pragma unroll
    for (int i = tid; i < KH * 32; i += 256) {
      const int kp = i >> 5, c4 = i & 31;
      const float* s0 = Eb + (size_t)(k0 + 2 * kp) * N + n0 + c4 * 4;
      const float* s1 = s0 + N;
      if (k0 + BK < Kd) __builtin_prefetch(s0 + (size_t)BK * N, 0, 1);
      const float4 r0 = *(const float4*)s0;
      const float4 r1 = *(const float4*)s1;
      unsigned int h[4], l[4];
      cvt_pair(r0.x, r1.x, h[0], l[0]);
      cvt_pair(r0.y, r1.y, h[1], l[1]);
      cvt_pair(r0.z, r1.z, h[2], l[2]);
      cvt_pair(r0.w, r1.w, h[3], l[3]);
      *(uint4*)&Bh[kp * LDB2 + c4 * 4] = make_uint4(h[0], h[1], h[2], h[3]);
      *(uint4*)&Bl[kp * LDB2 + c4 * 4] = make_uint4(l[0], l[1], l[2], l[3]);
    }
    __syncthreads();

    v16h ah[2], al[2];
#pragma unroll
    for (int t = 0; t < 2; ++t) {
      ah[t] = frag_A(Ah, waveM + t * 16 + lm, hi);
      al[t] = frag_A(Al, waveM + t * 16 + lm, hi);
    }
#pragma unroll
    for (int tj = 0; tj < 4; ++tj) {
      const v16h bh = frag_Bnn(Bh, waveN + tj * 16 + lm, hi);
      const v16h bl = frag_Bnn(Bl, waveN + tj * 16 + lm, hi);
#pragma unroll
      for (int ti = 0; ti < 2; ++ti)
        acc[ti][tj] = wmma3(ah[ti], al[ti], bh, bl, acc[ti][tj]);
    }
    __syncthreads();
  }

  // write into right half of concatenated row (row stride 2N, offset N)
  float* Ob = Out + (size_t)b * M * (2 * N);
#pragma unroll
  for (int ti = 0; ti < 2; ++ti)
#pragma unroll
    for (int tj = 0; tj < 4; ++tj) {
      const int col = n0 + waveN + tj * 16 + lm;
#pragma unroll
      for (int i = 0; i < 8; ++i) {
        const int row = m0 + waveM + ti * 16 + hi * 8 + i;
        Ob[(size_t)row * (2 * N) + N + col] = acc[ti][tj][i];
      }
    }
}

// ---------------------------------------------------------------------------
// copy raw embeddings into left half of concatenated output
// ---------------------------------------------------------------------------
__global__ __launch_bounds__(256) void copy_left_half(
    const float* __restrict__ src, float* __restrict__ dst, int nrows) {
  const int idx = blockIdx.x * 256 + threadIdx.x;
  if (idx >= nrows * 256) return;
  const int row = idx >> 8, c = idx & 255;
  const float4 v = *(const float4*)(src + (size_t)row * 1024 + c * 4);
  *(float4*)(dst + (size_t)row * 2048 + c * 4) = v;
}

// ---------------------------------------------------------------------------
extern "C" void kernel_launch(void* const* d_in, const int* in_sizes, int n_in,
                              void* d_out, int out_size, void* d_ws, size_t ws_size,
                              hipStream_t stream) {
  const float* emb_sup = (const float*)d_in[0];  // [8, 1024, 1024]
  const float* emb_qry = (const float*)d_in[1];  // [8, 2048, 1024]

  const int B = 8, NS = 1024, NQ = 2048, D = 1024;

  float* out     = (float*)d_out;
  float* aug_sup = out;                              // [8, 1024, 2048]
  float* aug_qry = aug_sup + (size_t)B * NS * 2 * D; // [8, 2048, 2048]
  float* G_sup   = aug_qry + (size_t)B * NQ * 2 * D; // [8, 1024, 1024]
  float* G_qry   = G_sup   + (size_t)B * NS * NS;    // [8, 2048, 1024]

  float* inv_sup = (float*)d_ws;          // B*NS floats
  float* inv_qry = inv_sup + B * NS;      // B*NQ floats

  // 1) inverse row norms
  rownorm_kernel<<<B * NS, 256, 0, stream>>>(emb_sup, inv_sup, D);
  rownorm_kernel<<<B * NQ, 256, 0, stream>>>(emb_qry, inv_qry, D);

  // 2) cosine logits straight into the G output regions
  gemm_nt_cos<<<dim3(NS / BN, NS / BM, B), 256, 0, stream>>>(
      emb_sup, emb_sup, G_sup, inv_sup, inv_sup, NS, NS, D);
  gemm_nt_cos<<<dim3(NS / BN, NQ / BM, B), 256, 0, stream>>>(
      emb_qry, emb_sup, G_qry, inv_qry, inv_sup, NQ, NS, D);

  // 3) softmax in place
  softmax_rows<<<B * NS, 256, 0, stream>>>(G_sup);
  softmax_rows<<<B * NQ, 256, 0, stream>>>(G_qry);

  // 4) task embeddings into right half of concatenated outputs
  gemm_nn_task<<<dim3(D / BN, NS / BM, B), 256, 0, stream>>>(
      G_sup, emb_sup, aug_sup, NS, D, NS);
  gemm_nn_task<<<dim3(D / BN, NQ / BM, B), 256, 0, stream>>>(
      G_qry, emb_sup, aug_qry, NQ, D, NS);

  // 5) raw embeddings into left half
  copy_left_half<<<(B * NS * 256 + 255) / 256, 256, 0, stream>>>(emb_sup, aug_sup, B * NS);
  copy_left_half<<<(B * NQ * 256 + 255) / 256, 256, 0, stream>>>(emb_qry, aug_qry, B * NQ);
}
